// Encoder_41669772706622
// MI455X (gfx1250) — matrix operand order repaired and checked by reading, not compiled, not run on hardware
//
#include <hip/hip_runtime.h>

typedef _Float16 v16h __attribute__((ext_vector_type(16)));
typedef float    v8f  __attribute__((ext_vector_type(8)));
typedef _Float16 h2v  __attribute__((ext_vector_type(2)));

#define HID 128

__device__ __forceinline__ unsigned pkadd_u(unsigned a, unsigned b) {
  h2v x = __builtin_bit_cast(h2v, a);
  h2v y = __builtin_bit_cast(h2v, b);
  h2v r = x + y;                       // v_pk_add_f16
  return __builtin_bit_cast(unsigned, r);
}

__device__ __forceinline__ void atomic_pk_add(_Float16* addr, unsigned val) {
#if __has_builtin(__builtin_amdgcn_global_atomic_fadd_v2f16)
  __builtin_amdgcn_global_atomic_fadd_v2f16(
      (__attribute__((address_space(1))) h2v*)addr, __builtin_bit_cast(h2v, val));
#else
  asm volatile("global_atomic_pk_add_f16 %0, %1, off"
               :: "v"(addr), "v"(val) : "memory");
#endif
}

// ---------------- embedding: x_f16[n][c] = emb[tok[n]][c] ----------------
__global__ void enc_kernel(const int* __restrict__ tok,
                           const float* __restrict__ emb,
                           _Float16* __restrict__ x, int N) {
  int tid = blockIdx.x * blockDim.x + threadIdx.x;
  int node = tid >> 5;
  int c = (tid & 31) * 4;
  if (node >= N) return;
  float4 v = *(const float4*)(emb + (size_t)tok[node] * HID + c);
  h2v lo = { (_Float16)v.x, (_Float16)v.y };
  h2v hi = { (_Float16)v.z, (_Float16)v.w };
  uint2 o;
  o.x = __builtin_bit_cast(unsigned, lo);
  o.y = __builtin_bit_cast(unsigned, hi);
  *(uint2*)(x + (size_t)node * HID + c) = o;
}

// ---------------- edge scatter-add: agg[dst] += x[src] (pk f16 atomics) ----
__global__ void scatter_kernel(const _Float16* __restrict__ x,
                               const int* __restrict__ src,
                               const int* __restrict__ dst,
                               _Float16* __restrict__ agg, int E) {
  long long tid = (long long)blockIdx.x * blockDim.x + threadIdx.x;
  int e = (int)(tid >> 3);
  if (e >= E) return;
  int p = ((int)tid & 7) * 16;          // 16 halves (32B) per thread
  int s = src[e], d = dst[e];
  const uint4* xs = (const uint4*)(x + (size_t)s * HID + p);
  uint4 v0 = xs[0], v1 = xs[1];
  _Float16* ap = agg + (size_t)d * HID + p;
  const unsigned* u0 = (const unsigned*)&v0;
  const unsigned* u1 = (const unsigned*)&v1;
#pragma unroll
  for (int j = 0; j < 4; ++j) atomic_pk_add(ap + 2 * j, u0[j]);
#pragma unroll
  for (int j = 0; j < 4; ++j) atomic_pk_add(ap + 8 + 2 * j, u1[j]);
}

// ---------------- fused GEMM: out = relu((A [+Agg]) @ W + b), f16 WMMA ----
// one wave computes a 16-row x 128-col strip; W staged in LDS (f16,
// transposed, XOR-swizzled 16B units for conflict-free ds_load_b128).
template <bool ADD>
__global__ __launch_bounds__(256) void gin_gemm(
    const _Float16* __restrict__ A, const _Float16* __restrict__ Agg,
    const float* __restrict__ W, const float* __restrict__ bias,
    _Float16* __restrict__ out, int nRows, int nStrips) {
  __shared__ __align__(16) unsigned short WTs[HID * HID];  // 32 KB

  int t = threadIdx.x;
  // cooperative stage: WTs[n][swizzle(k)] = (f16) W[k][n]
#pragma unroll
  for (int i = 0; i < 16; ++i) {
    int idx4 = t + 256 * i;                 // 4096 float4 total
    float4 w4 = *(const float4*)(W + (size_t)idx4 * 4);
    int k = idx4 >> 5;
    int n0 = (idx4 & 31) * 4;
    const float* wf = &w4.x;
#pragma unroll
    for (int e = 0; e < 4; ++e) {
      int n = n0 + e;
      _Float16 hv = (_Float16)wf[e];
      WTs[n * HID + (((k >> 3) ^ (n & 15)) << 3) + (k & 7)] =
          __builtin_bit_cast(unsigned short, hv);
    }
  }
  __syncthreads();

  int wave = t >> 5;
  int lane = t & 31;
  int h = lane >> 4;        // half-wave
  int mn = lane & 15;       // row (A) / col (B,D) within tile
  int s = blockIdx.x * 8 + wave;
  if (s >= nStrips) return;
  int r0 = s * 16;

  int arow = r0 + mn;
  if (arow >= nRows) arow = nRows - 1;
  const _Float16* Ap = A + (size_t)arow * HID;
  const _Float16* Gp = Agg + (size_t)arow * HID;

  v8f acc[8];
#pragma unroll
  for (int nt = 0; nt < 8; ++nt) {
    float b = bias[nt * 16 + mn];
    v8f a;
#pragma unroll
    for (int r = 0; r < 8; ++r) a[r] = b;
    acc[nt] = a;
  }

#pragma unroll
  for (int kt = 0; kt < 4; ++kt) {
    // A fragment: lane(h,m) holds K = 32kt+8h+j (j<8) and 32kt+16+8h+j
    union { v16h v; uint4 q[2]; unsigned u[8]; } af;
    af.q[0] = *(const uint4*)(Ap + kt * 32 + 8 * h);
    af.q[1] = *(const uint4*)(Ap + kt * 32 + 16 + 8 * h);
    if (ADD) {
      uint4 g0 = *(const uint4*)(Gp + kt * 32 + 8 * h);
      uint4 g1 = *(const uint4*)(Gp + kt * 32 + 16 + 8 * h);
      const unsigned* gu0 = (const unsigned*)&g0;
      const unsigned* gu1 = (const unsigned*)&g1;
#pragma unroll
      for (int j = 0; j < 4; ++j) af.u[j] = pkadd_u(af.u[j], gu0[j]);
#pragma unroll
      for (int j = 0; j < 4; ++j) af.u[4 + j] = pkadd_u(af.u[4 + j], gu1[j]);
    }
#pragma unroll
    for (int nt = 0; nt < 8; ++nt) {
      int row = nt * 16 + mn;
      int ub = 4 * kt + h;       // 16B-unit index along K
      union { v16h v; uint4 q[2]; } bf;
      bf.q[0] = *(const uint4*)(WTs + row * HID + ((ub ^ mn) << 3));
      bf.q[1] = *(const uint4*)(WTs + row * HID + (((ub + 2) ^ mn) << 3));
      acc[nt] = __builtin_amdgcn_wmma_f32_16x16x32_f16(
          false, af.v, false, bf.v, (short)0, acc[nt], false, false);
    }
  }

#pragma unroll
  for (int nt = 0; nt < 8; ++nt) {
#pragma unroll
    for (int r = 0; r < 8; ++r) {
      int row = r0 + 8 * h + r;      // D layout: VGPR r -> M = r + 8*half
      if (row < nRows) {
        float v = acc[nt][r];
        v = v > 0.f ? v : 0.f;       // ReLU (both MLP stages end in relu)
        out[(size_t)row * HID + nt * 16 + mn] = (_Float16)v;
      }
    }
  }
}

// ---------------- per-graph mean pool ----------------
__global__ void pool_accum(const _Float16* __restrict__ x,
                           const int* __restrict__ batch,
                           float* __restrict__ sums,
                           float* __restrict__ counts, int N) {
  int tid = blockIdx.x * blockDim.x + threadIdx.x;
  int node = tid >> 5;
  int c = (tid & 31) * 4;
  if (node >= N) return;
  int b = batch[node];
  uint2 v = *(const uint2*)(x + (size_t)node * HID + c);
  h2v lo = __builtin_bit_cast(h2v, v.x);
  h2v hi = __builtin_bit_cast(h2v, v.y);
  float* sp = sums + (size_t)b * HID + c;
  atomicAdd(sp + 0, (float)lo[0]);
  atomicAdd(sp + 1, (float)lo[1]);
  atomicAdd(sp + 2, (float)hi[0]);
  atomicAdd(sp + 3, (float)hi[1]);
  if ((tid & 31) == 0) atomicAdd(&counts[b], 1.0f);
}

__global__ void pool_final(const float* __restrict__ sums,
                           const float* __restrict__ counts,
                           float* __restrict__ out, int total) {
  int idx = blockIdx.x * blockDim.x + threadIdx.x;
  if (idx >= total) return;
  float cn = counts[idx >> 7];
  out[idx] = sums[idx] / fmaxf(cn, 1.0f);
}

// ---------------- host orchestration ----------------
extern "C" void kernel_launch(void* const* d_in, const int* in_sizes, int n_in,
                              void* d_out, int out_size, void* d_ws,
                              size_t ws_size, hipStream_t stream) {
  const int*   tok   = (const int*)d_in[0];
  const int*   eidx  = (const int*)d_in[1];
  const int*   batch = (const int*)d_in[2];
  const float* emb   = (const float*)d_in[3];
  const float* W1    = (const float*)d_in[4];
  const float* b1    = (const float*)d_in[5];
  const float* W2    = (const float*)d_in[6];
  const float* b2    = (const float*)d_in[7];

  const int N = in_sizes[0];
  const int E = in_sizes[1] / 2;
  const int LAYERS = 6;
  const int G = out_size / HID;

  const int* src = eidx;
  const int* dst = eidx + E;

  size_t xb = ((size_t)N * HID * sizeof(_Float16) + 255) & ~(size_t)255;
  char* ws = (char*)d_ws;
  _Float16* bufA = (_Float16*)(ws + 0 * xb);
  _Float16* bufB = (_Float16*)(ws + 1 * xb);
  _Float16* hmid = (_Float16*)(ws + 2 * xb);
  _Float16* agg  = (_Float16*)(ws + 3 * xb);
  float* sums    = (float*)(ws + 4 * xb);
  float* counts  = (float*)((char*)sums + ((size_t)G * HID * sizeof(float) + 255 & ~(size_t)255));

  const int BLK = 256;
  int nStrips = (N + 15) / 16;
  int gemmBlocks = (nStrips + 7) / 8;

  // embedding
  {
    long long th = (long long)N * 32;
    enc_kernel<<<(unsigned)((th + BLK - 1) / BLK), BLK, 0, stream>>>(tok, emb, bufA, N);
  }

  _Float16* cur = bufA;
  _Float16* nxt = bufB;
  for (int l = 0; l < LAYERS; ++l) {
    hipMemsetAsync(agg, 0, (size_t)N * HID * sizeof(_Float16), stream);
    {
      long long th = (long long)E * 8;
      scatter_kernel<<<(unsigned)((th + BLK - 1) / BLK), BLK, 0, stream>>>(cur, src, dst, agg, E);
    }
    gin_gemm<true><<<gemmBlocks, BLK, 0, stream>>>(
        cur, agg, W1 + (size_t)l * HID * HID, b1 + (size_t)l * HID, hmid, N, nStrips);
    gin_gemm<false><<<gemmBlocks, BLK, 0, stream>>>(
        hmid, agg, W2 + (size_t)l * HID * HID, b2 + (size_t)l * HID, nxt, N, nStrips);
    _Float16* t = cur; cur = nxt; nxt = t;
  }

  hipMemsetAsync(sums, 0, (size_t)G * HID * sizeof(float), stream);
  hipMemsetAsync(counts, 0, (size_t)G * sizeof(float), stream);
  {
    long long th = (long long)N * 32;
    pool_accum<<<(unsigned)((th + BLK - 1) / BLK), BLK, 0, stream>>>(cur, batch, sums, counts, N);
  }
  {
    int total = G * HID;
    pool_final<<<(total + BLK - 1) / BLK, BLK, 0, stream>>>(sums, counts, (float*)d_out, total);
  }
}
